// NodeProcessorModule_39298950758850
// MI455X (gfx1250) — matrix operand order repaired and checked by reading, not compiled, not run on hardware
//
#include <hip/hip_runtime.h>

typedef __attribute__((ext_vector_type(16))) __bf16 v16bf;
typedef __attribute__((ext_vector_type(8)))  float  v8f;
typedef unsigned short u16;
typedef unsigned int   u32;

#define FDIM 128
#define HDIM 512
#define TM   64   // nodes per block (4 M-tiles of 16)

// ---------- helpers ----------
__device__ __forceinline__ u16 f2bf(float f) {
  u32 u = __float_as_uint(f);
  u32 r = u + 0x7FFFu + ((u >> 16) & 1u);   // round-to-nearest-even
  return (u16)(r >> 16);
}

union Frag { v16bf v; uint4 q[2]; };

// A fragment (16x32 bf16) from LDS, ISA layout:
// lanes 0-15 : row M=lane,    K = kk+0..7  and kk+16..23
// lanes 16-31: row M=lane-16, K = kk+8..15 and kk+24..31
__device__ __forceinline__ v16bf load_afrag(const u16* sh, int ld, int lane, int kk) {
  Frag u;
  const int r  = lane & 15;
  const int kh = (lane >> 4) << 3;
  const u16* p = sh + r * ld + kk + kh;
  u.q[0] = *(const uint4*)(p);        // ds_load_b128
  u.q[1] = *(const uint4*)(p + 16);
  return u.v;
}

// B fragment: pre-packed so each lane reads 16 contiguous bf16 (32B)
__device__ __forceinline__ v16bf load_bfrag(const u16* __restrict__ Wp, int frag, int lane) {
  Frag u;
  const uint4* p = (const uint4*)(Wp + ((size_t)frag * 32 + lane) * 16);
  u.q[0] = p[0];
  u.q[1] = p[1];
  return u.v;
}

// One GEMM layer on a 64-node tile. 4 waves; each wave owns N-tile pairs
// {nt, nt+4} (stride 8) and all four 16-row M-tiles: 4x2 register blocking,
// so every B fragment feeds 4 WMMAs and every A fragment feeds 2 WMMAs.
template<bool FINAL>
__device__ __forceinline__ void gemm_layer(
    const u16* src, int srcLd, int K,
    const u16* __restrict__ Wp, const float* __restrict__ bias, int NT,
    u16* dst, int dstLd, float* fdst, int fLd,
    int lane, int wave)
{
  const int KT  = K >> 5;
  const int nlo = lane & 15;

  for (int nt = wave; nt < NT; nt += 8) {
    const int nt2 = nt + 4;
    const float bv0 = bias[nt  * 16 + nlo];
    const float bv1 = bias[nt2 * 16 + nlo];
    v8f acc[4][2];                               // acc[mtile][ntile]
#pragma unroll
    for (int mt = 0; mt < 4; ++mt)
#pragma unroll
      for (int i = 0; i < 8; ++i) { acc[mt][0][i] = bv0; acc[mt][1][i] = bv1; }

    for (int kt = 0; kt < KT; ++kt) {
      v16bf w0 = load_bfrag(Wp, nt  * KT + kt, lane);
      v16bf w1 = load_bfrag(Wp, nt2 * KT + kt, lane);
#pragma unroll
      for (int mt = 0; mt < 4; ++mt) {
        v16bf a = load_afrag(src + mt * 16 * srcLd, srcLd, lane, kt << 5);
        acc[mt][0] = __builtin_amdgcn_wmma_f32_16x16x32_bf16(false, a, false, w0,
                                                             (short)0, acc[mt][0], false, false);
        acc[mt][1] = __builtin_amdgcn_wmma_f32_16x16x32_bf16(false, a, false, w1,
                                                             (short)0, acc[mt][1], false, false);
      }
    }

    // C/D layout: VGPR r -> M = r (+8 for lanes 16-31), N = lane&15
    const int mb = (lane >> 4) << 3;
#pragma unroll
    for (int mt = 0; mt < 4; ++mt) {
#pragma unroll
      for (int r = 0; r < 8; ++r) {
        const int m = mt * 16 + mb + r;
        if (FINAL) {
          fdst[m * fLd + nt  * 16 + nlo] = acc[mt][0][r];
          fdst[m * fLd + nt2 * 16 + nlo] = acc[mt][1][r];
        } else {
          float x0 = acc[mt][0][r]; x0 = x0 > 0.f ? x0 : 0.f;   // ReLU
          float x1 = acc[mt][1][r]; x1 = x1 > 0.f ? x1 : 0.f;
          dst[m * dstLd + nt  * 16 + nlo] = f2bf(x0);
          dst[m * dstLd + nt2 * 16 + nlo] = f2bf(x1);
        }
      }
    }
  }
}

// ---------- kernels ----------
__global__ void zero_kernel(float* __restrict__ p, int n) {
  int i = blockIdx.x * blockDim.x + threadIdx.x;
  if (i < n) p[i] = 0.f;
}

// one thread per (edge, 4 features): float4 load + 4 L2 atomics
__global__ void scatter_kernel(const float* __restrict__ edge,
                               const int* __restrict__ recv,
                               float* __restrict__ agg, int nEdges) {
  int t = blockIdx.x * blockDim.x + threadIdx.x;
  int e = t >> 5;
  if (e >= nEdges) return;
  int c = (t & 31) * 4;
  float4 v = *(const float4*)(edge + (size_t)e * FDIM + c);
  float* dst = agg + (size_t)recv[e] * FDIM + c;
  atomicAdd(dst + 0, v.x);
  atomicAdd(dst + 1, v.y);
  atomicAdd(dst + 2, v.z);
  atomicAdd(dst + 3, v.w);
}

// pack W[K][N] (row-major f32) into per-lane-contiguous bf16 B fragments
__global__ void pack_kernel(const float* __restrict__ W, u16* __restrict__ Wp,
                            int K, int N) {
  const int KT = K >> 5;
  const int total = K * N;
  for (int idx = blockIdx.x * blockDim.x + threadIdx.x; idx < total;
       idx += gridDim.x * blockDim.x) {
    int i  = idx & 15;
    int l  = (idx >> 4) & 31;
    int f  = idx >> 9;
    int kt = f % KT;
    int nt = f / KT;
    int k  = kt * 32 + ((l >> 4) << 3) + (i & 7) + ((i >> 3) << 4);  // lanes 0-15: K 0..7,16..23; 16-31: K 8..15,24..31
    int n  = nt * 16 + (l & 15);
    Wp[idx] = f2bf(W[(size_t)k * N + n]);
  }
}

__global__ __launch_bounds__(128) void mlp_kernel(
    const float* __restrict__ node_attr, const float* __restrict__ agg,
    const u16* __restrict__ W1p, const float* __restrict__ b1,
    const u16* __restrict__ W2p, const float* __restrict__ b2,
    const u16* __restrict__ W3p, const float* __restrict__ b3,
    const float* __restrict__ gamma, const float* __restrict__ beta,
    float* __restrict__ out, int nNodes)
{
  __shared__ u16  bufX[TM * HDIM];   // 64 KB: input (ld=256), later h2 (ld=512)
  __shared__ u16  bufY[TM * HDIM];   // 64 KB: h1
  __shared__ float sOut[TM * FDIM];  // 32 KB: layer-3 f32 output

  const int tid  = threadIdx.x;
  const int lane = tid & 31;
  const int wave = tid >> 5;
  const int node0 = blockIdx.x * TM;

  // stage x = [node_attr | agg] -> bf16 in LDS, row stride 256
  for (int idx = tid; idx < TM * 64; idx += 128) {   // 64 float4 per row
    int r = idx >> 6;
    int c = (idx & 63) * 4;
    float4 v = {0.f, 0.f, 0.f, 0.f};
    int row = node0 + r;
    if (row < nNodes) {
      if (c < FDIM) v = *(const float4*)(node_attr + (size_t)row * FDIM + c);
      else          v = *(const float4*)(agg + (size_t)row * FDIM + (c - FDIM));
    }
    u16* p = bufX + r * 256 + c;
    p[0] = f2bf(v.x); p[1] = f2bf(v.y); p[2] = f2bf(v.z); p[3] = f2bf(v.w);
  }
  __syncthreads();

  gemm_layer<false>(bufX, 256, 256, W1p, b1, HDIM / 16, bufY, HDIM, nullptr, 0, lane, wave);
  __syncthreads();
  gemm_layer<false>(bufY, HDIM, HDIM, W2p, b2, HDIM / 16, bufX, HDIM, nullptr, 0, lane, wave);
  __syncthreads();
  gemm_layer<true >(bufX, HDIM, HDIM, W3p, b3, FDIM / 16, nullptr, 0, sOut, FDIM, lane, wave);
  __syncthreads();

  // LayerNorm over F=128 per node: 2 lanes per row, shuffle reduce (wave32)
  const int row = tid >> 1;          // 0..63
  const int sub = tid & 1;
  float s = 0.f, s2 = 0.f;
  for (int c = sub; c < FDIM; c += 2) {
    float v = sOut[row * FDIM + c];
    s += v; s2 += v * v;
  }
  s  += __shfl_xor(s,  1, 32);
  s2 += __shfl_xor(s2, 1, 32);
  const float mu  = s * (1.f / FDIM);
  const float var = s2 * (1.f / FDIM) - mu * mu;
  const float rs  = rsqrtf(var + 1e-5f);
  const int grow = node0 + row;
  if (grow < nNodes) {
    for (int c = sub; c < FDIM; c += 2) {
      out[(size_t)grow * FDIM + c] =
          (sOut[row * FDIM + c] - mu) * rs * gamma[c] + beta[c];
    }
  }
}

// ---------- launcher ----------
extern "C" void kernel_launch(void* const* d_in, const int* in_sizes, int n_in,
                              void* d_out, int out_size, void* d_ws, size_t ws_size,
                              hipStream_t stream) {
  const float* node_attr = (const float*)d_in[0];
  const float* edge_attr = (const float*)d_in[1];
  const int*   recv      = (const int*)d_in[2];
  const float* W1 = (const float*)d_in[3];
  const float* b1 = (const float*)d_in[4];
  const float* W2 = (const float*)d_in[5];
  const float* b2 = (const float*)d_in[6];
  const float* W3 = (const float*)d_in[7];
  const float* b3 = (const float*)d_in[8];
  const float* gamma = (const float*)d_in[9];
  const float* beta  = (const float*)d_in[10];
  float* out = (float*)d_out;

  const int nNodes = in_sizes[0] / FDIM;
  const int nEdges = in_sizes[1] / FDIM;

  // workspace layout: agg (f32) | W1p | W2p | W3p (bf16 fragments)
  float* agg = (float*)d_ws;
  size_t aggBytes = ((size_t)nNodes * FDIM * sizeof(float) + 255) & ~(size_t)255;
  u16* W1p = (u16*)((char*)d_ws + aggBytes);
  u16* W2p = W1p + 2 * FDIM * HDIM;
  u16* W3p = W2p + HDIM * HDIM;

  const int aggN = nNodes * FDIM;
  zero_kernel<<<(aggN + 255) / 256, 256, 0, stream>>>(agg, aggN);

  const int scatterThreads = nEdges * 32;
  scatter_kernel<<<(scatterThreads + 255) / 256, 256, 0, stream>>>(
      edge_attr, recv, agg, nEdges);

  pack_kernel<<<(2 * FDIM * HDIM + 255) / 256, 256, 0, stream>>>(W1, W1p, 2 * FDIM, HDIM);
  pack_kernel<<<(HDIM * HDIM + 255) / 256, 256, 0, stream>>>(W2, W2p, HDIM, HDIM);
  pack_kernel<<<(HDIM * FDIM + 255) / 256, 256, 0, stream>>>(W3, W3p, HDIM, FDIM);

  mlp_kernel<<<(nNodes + TM - 1) / TM, 128, 0, stream>>>(
      node_attr, agg, W1p, b1, W2p, b2, W3p, b3, gamma, beta, out, nNodes);
}
